// Attn_head_9543417332154
// MI455X (gfx1250) — matrix operand order repaired and checked by reading, not compiled, not run on hardware
//
#include <hip/hip_runtime.h>
#include <stdint.h>

typedef __attribute__((ext_vector_type(16))) _Float16 v16h;
typedef __attribute__((ext_vector_type(8)))  float    v8f;

#define N_NODES 8192
#define C_IN    256
#define O_OUT   64
#define F16_SCALE 4096.0f

#if __has_builtin(__builtin_amdgcn_global_load_async_to_lds_b128)
#define USE_ASYNC_LDS 1
#else
#define USE_ASYNC_LDS 0
#endif

static __device__ __forceinline__ void async_cp_b128(const void* g, const void* l) {
#if USE_ASYNC_LDS
  typedef int v4i __attribute__((vector_size(4 * sizeof(int))));
  __builtin_amdgcn_global_load_async_to_lds_b128(
      (__attribute__((address_space(1))) v4i*)(uintptr_t)g,
      (__attribute__((address_space(3))) v4i*)(uintptr_t)l,
      0, 0);
#else
  (void)g; (void)l;
#endif
}

template <int N>
static __device__ __forceinline__ void wait_asynccnt() {
#if __has_builtin(__builtin_amdgcn_s_wait_asynccnt)
  __builtin_amdgcn_s_wait_asynccnt((unsigned short)N);
#elif USE_ASYNC_LDS
  asm volatile("s_wait_asynccnt %0" :: "i"(N) : "memory");
#endif
}

static __device__ __forceinline__ v16h cvt16(float4 a, float4 b, float4 c, float4 d) {
  v16h r;
  r[0]  = (_Float16)a.x; r[1]  = (_Float16)a.y; r[2]  = (_Float16)a.z; r[3]  = (_Float16)a.w;
  r[4]  = (_Float16)b.x; r[5]  = (_Float16)b.y; r[6]  = (_Float16)b.z; r[7]  = (_Float16)b.w;
  r[8]  = (_Float16)c.x; r[9]  = (_Float16)c.y; r[10] = (_Float16)c.z; r[11] = (_Float16)c.w;
  r[12] = (_Float16)d.x; r[13] = (_Float16)d.y; r[14] = (_Float16)d.z; r[15] = (_Float16)d.w;
  return r;
}

// ---------------------------------------------------------------------------
// K1: seq_fts[n][o] = sum_c x[n][c] * W1[o][c] + b1[o]   (WMMA f16, f32 acc)
// ---------------------------------------------------------------------------
__global__ void k1_seqfts(const float* __restrict__ x, const float* __restrict__ W1,
                          const float* __restrict__ b1, float* __restrict__ sf) {
  const int wave = threadIdx.x >> 5;
  const int lane = threadIdx.x & 31;
  const int hl   = lane >> 4;
  const int mr   = lane & 15;
  const int row  = blockIdx.x * 16 + mr;   // A row (n)
  const int ocol = wave * 16 + mr;         // B col (o)
  const float* xrow = x  + (size_t)row  * C_IN;
  const float* wrow = W1 + (size_t)ocol * C_IN;
  v8f acc = {0.f,0.f,0.f,0.f,0.f,0.f,0.f,0.f};
#pragma unroll
  for (int kc = 0; kc < 8; ++kc) {
    const int kb = kc * 32;
    const float4* ap0 = reinterpret_cast<const float4*>(xrow + kb + hl * 8);
    const float4* ap1 = reinterpret_cast<const float4*>(xrow + kb + 16 + hl * 8);
    v16h A = cvt16(ap0[0], ap0[1], ap1[0], ap1[1]);
    const float4* bp0 = reinterpret_cast<const float4*>(wrow + kb + hl * 8);
    const float4* bp1 = reinterpret_cast<const float4*>(wrow + kb + 16 + hl * 8);
    v16h Bv = cvt16(bp0[0], bp0[1], bp1[0], bp1[1]);
    acc = __builtin_amdgcn_wmma_f32_16x16x32_f16(false, A, false, Bv, (short)0, acc, false, false);
  }
  const float bias = b1[ocol];
#pragma unroll
  for (int r = 0; r < 8; ++r) {
    const int m = r + hl * 8;
    sf[(size_t)(blockIdx.x * 16 + m) * O_OUT + ocol] = acc[r] + bias;
  }
}

// ---------------------------------------------------------------------------
// K2: f1[n], f2[n] (64-dot) + per-block max of f1
// ---------------------------------------------------------------------------
__global__ void k2_fts(const float* __restrict__ sf, const float* __restrict__ a1,
                       const float* __restrict__ ba1, const float* __restrict__ a2,
                       const float* __restrict__ ba2, float* __restrict__ f1,
                       float* __restrict__ f2, float* __restrict__ pmax) {
  const int n = blockIdx.x * 256 + threadIdx.x;
  const float4* rowp = reinterpret_cast<const float4*>(sf + (size_t)n * O_OUT);
  const float4* va = reinterpret_cast<const float4*>(a1);
  const float4* vb = reinterpret_cast<const float4*>(a2);
  float s1 = 0.f, s2 = 0.f;
#pragma unroll
  for (int q = 0; q < 16; ++q) {
    float4 v = rowp[q]; float4 c1 = va[q]; float4 c2 = vb[q];
    s1 += v.x*c1.x + v.y*c1.y + v.z*c1.z + v.w*c1.w;
    s2 += v.x*c2.x + v.y*c2.y + v.z*c2.z + v.w*c2.w;
  }
  s1 += ba1[0]; s2 += ba2[0];
  f1[n] = s1; f2[n] = s2;
  __shared__ float red[256];
  red[threadIdx.x] = s1;
  __syncthreads();
  for (int off = 128; off > 0; off >>= 1) {
    if (threadIdx.x < off) red[threadIdx.x] = fmaxf(red[threadIdx.x], red[threadIdx.x + off]);
    __syncthreads();
  }
  if (threadIdx.x == 0) pmax[blockIdx.x] = red[0];
}

// ---------------------------------------------------------------------------
// K3: f1max, m[j] = lrelu(f1max + f2[j]); E1 = exp(f1), E2 = exp(0.01 f1); zero W
// ---------------------------------------------------------------------------
__global__ void k3_prep(const float* __restrict__ pmax, const float* __restrict__ f1,
                        const float* __restrict__ f2, float* __restrict__ m,
                        float* __restrict__ E1, float* __restrict__ E2,
                        float* __restrict__ Wc) {
  __shared__ float red[33];
  if (threadIdx.x < 32) red[threadIdx.x] = pmax[threadIdx.x];
  __syncthreads();
  if (threadIdx.x == 0) {
    float mx = red[0];
    for (int i = 1; i < 32; ++i) mx = fmaxf(mx, red[i]);
    red[32] = mx;
  }
  __syncthreads();
  const float f1max = red[32];
  if (threadIdx.x < O_OUT) Wc[threadIdx.x] = 0.0f;
  for (int j = threadIdx.x; j < N_NODES; j += 256) {
    const float s = f1max + f2[j];
    m[j] = s > 0.0f ? s : 0.01f * s;
    const float v = f1[j];
    E1[j] = __expf(v);
    E2[j] = __expf(0.01f * v);
  }
}

// ---------------------------------------------------------------------------
// K4: Z[j] = e^{f2j-mj} * sum_{f1i+f2j>0} E1[i] + e^{0.01 f2j-mj} * sum_{<=0} E2[i]
// ---------------------------------------------------------------------------
__global__ void k4_z(const float* __restrict__ f1, const float* __restrict__ E1,
                     const float* __restrict__ E2, const float* __restrict__ f2,
                     const float* __restrict__ m, float* __restrict__ Z) {
  const int j = blockIdx.x * 256 + threadIdx.x;
  const float thr = -f2[j];
  float accA = 0.f, accB = 0.f;
  for (int i = 0; i < N_NODES; i += 4) {
    float4 fv = *reinterpret_cast<const float4*>(f1 + i);
    float4 e1 = *reinterpret_cast<const float4*>(E1 + i);
    float4 e2 = *reinterpret_cast<const float4*>(E2 + i);
    accA += (fv.x > thr) ? e1.x : 0.f;  accB += (fv.x > thr) ? 0.f : e2.x;
    accA += (fv.y > thr) ? e1.y : 0.f;  accB += (fv.y > thr) ? 0.f : e2.y;
    accA += (fv.z > thr) ? e1.z : 0.f;  accB += (fv.z > thr) ? 0.f : e2.z;
    accA += (fv.w > thr) ? e1.w : 0.f;  accB += (fv.w > thr) ? 0.f : e2.w;
  }
  const float mj = m[j], f2j = f2[j];
  Z[j] = __expf(f2j - mj) * accA + __expf(0.01f * f2j - mj) * accB;
}

// ---------------------------------------------------------------------------
// K5: pack u/w into WMMA-B-tile layout f16 chunks; column-sum of w into Wc.
// ---------------------------------------------------------------------------
__global__ void k5_pack(const float* __restrict__ sf, const float* __restrict__ f2,
                        const float* __restrict__ m, const float* __restrict__ Z,
                        _Float16* __restrict__ Bpk, float* __restrict__ Wc) {
  const int idx = blockIdx.x * 256 + threadIdx.x;
  const int j = idx >> 6, o = idx & 63;
  const float sfv = sf[(size_t)j * O_OUT + o];
  const float mj = m[j], f2j = f2[j];
  const float inv = F16_SCALE / Z[j];
  const float u = __expf(f2j - mj) * sfv * inv;
  const float w = __expf(0.01f * f2j - mj) * sfv * inv;
  const int chunk = j >> 5, kl = j & 31, halfk = kl >> 4, e = kl & 15;
  const int otile = o >> 4;
  const int lanep = halfk * 16 + (o & 15);
  const size_t base = (size_t)chunk * 4096 + otile * 512 + lanep * 16 + e;
  Bpk[base]        = (_Float16)u;
  Bpk[base + 2048] = (_Float16)w;
  __shared__ float red[256];
  red[threadIdx.x] = w;
  __syncthreads();
  if (threadIdx.x < 64) {
    float s = red[threadIdx.x] + red[threadIdx.x + 64] + red[threadIdx.x + 128] + red[threadIdx.x + 192];
    atomicAdd(&Wc[threadIdx.x], s);
  }
}

// ---------------------------------------------------------------------------
// K6: main WMMA loop. Block = 256 thr = 8 waves, each wave owns 16 i-rows.
// A = binary mask M[i,j]=[f1i+f2j>0] built in-register (f16 0/1).
// B tiles (u,w) double-buffered in LDS via GLOBAL_LOAD_ASYNC_TO_LDS_B128
// (ASYNCcnt-throttled) when available; sync staging fallback otherwise.
// ---------------------------------------------------------------------------
__global__ void k6_attn(const _Float16* __restrict__ Bpk, const float* __restrict__ f1,
                        const float* __restrict__ f2, const float* __restrict__ E1,
                        const float* __restrict__ E2, const float* __restrict__ Wc,
                        float* __restrict__ out) {
#if USE_ASYNC_LDS
  __shared__ _Float16 lds[2][4096] __attribute__((aligned(32)));
#else
  __shared__ _Float16 lds[1][4096] __attribute__((aligned(32)));
#endif
  const int t = threadIdx.x;
  const int wave = t >> 5;
  const int lane = t & 31;
  const int hl = lane >> 4;
  const int mr = lane & 15;
  const int ibase = blockIdx.x * 128 + wave * 16;
  const float f1v = f1[ibase + mr];
  const v8f vzero = {0.f,0.f,0.f,0.f,0.f,0.f,0.f,0.f};
  v8f c1[4], c2[4];
#pragma unroll
  for (int q = 0; q < 4; ++q) { c1[q] = vzero; c2[q] = vzero; }

#if USE_ASYNC_LDS
  // Prologue: async-stage chunk 0 into buffer 0 (2 x b128 per thread).
  async_cp_b128(Bpk + (size_t)t * 8,         &lds[0][t * 8]);
  async_cp_b128(Bpk + (size_t)(t + 256) * 8, &lds[0][(t + 256) * 8]);
#endif

  for (int jc = 0; jc < 256; ++jc) {
    const _Float16* Lb;
#if USE_ASYNC_LDS
    const int cur = jc & 1;
    __syncthreads();                       // prev compute done; safe to refill other buffer
    if (jc + 1 < 256) {
      const _Float16* g = Bpk + (size_t)(jc + 1) * 4096;
      async_cp_b128(g + (size_t)t * 8,         &lds[cur ^ 1][t * 8]);
      async_cp_b128(g + (size_t)(t + 256) * 8, &lds[cur ^ 1][(t + 256) * 8]);
      wait_asynccnt<2>();                  // only the 2 just-issued may remain outstanding
    } else {
      wait_asynccnt<0>();
    }
    __syncthreads();                       // chunk jc resident for all waves
    Lb = &lds[cur][0];
#else
    __syncthreads();
    {
      const float4* src = reinterpret_cast<const float4*>(Bpk + (size_t)jc * 4096);
      float4* dst = reinterpret_cast<float4*>(&lds[0][0]);
      dst[t] = src[t];
      dst[t + 256] = src[t + 256];
      __builtin_prefetch(Bpk + (size_t)(jc + 1) * 4096, 0, 0);
    }
    __syncthreads();
    Lb = &lds[0][0];
#endif

    const int jb = jc * 32;
    const float4* fp0 = reinterpret_cast<const float4*>(f2 + jb + hl * 16);
    const float4* fp1 = reinterpret_cast<const float4*>(f2 + jb + hl * 16 + 8);
    float4 p0 = fp0[0], p1 = fp0[1], p2 = fp1[0], p3 = fp1[1];
    const float fv[16] = {p0.x,p0.y,p0.z,p0.w, p1.x,p1.y,p1.z,p1.w,
                          p2.x,p2.y,p2.z,p2.w, p3.x,p3.y,p3.z,p3.w};
    v16h Am;
#pragma unroll
    for (int e = 0; e < 16; ++e)
      Am[e] = (_Float16)(((f1v + fv[e]) > 0.0f) ? 1.0f : 0.0f);

#pragma unroll
    for (int ot = 0; ot < 4; ++ot) {
      const v16h Bu = *reinterpret_cast<const v16h*>(Lb + ot * 512 + lane * 16);
      c1[ot] = __builtin_amdgcn_wmma_f32_16x16x32_f16(false, Am, false, Bu, (short)0, c1[ot], false, false);
      const v16h Bw = *reinterpret_cast<const v16h*>(Lb + 2048 + ot * 512 + lane * 16);
      c2[ot] = __builtin_amdgcn_wmma_f32_16x16x32_f16(false, Am, false, Bw, (short)0, c2[ot], false, false);
    }
  }

  float e1v[8], e2v[8];
#pragma unroll
  for (int r = 0; r < 8; ++r) {
    const int i = ibase + r + hl * 8;
    e1v[r] = E1[i]; e2v[r] = E2[i];
  }
#pragma unroll
  for (int ot = 0; ot < 4; ++ot) {
    const int o = ot * 16 + mr;
    const float Wo = Wc[o];
#pragma unroll
    for (int r = 0; r < 8; ++r) {
      const int i = ibase + r + hl * 8;
      float val = (e1v[r] * c1[ot][r] + e2v[r] * (Wo - c2[ot][r])) * (1.0f / F16_SCALE);
      out[(size_t)i * O_OUT + o] = val > 0.0f ? val : (__expf(val) - 1.0f);
    }
  }
}

// ---------------------------------------------------------------------------
extern "C" void kernel_launch(void* const* d_in, const int* in_sizes, int n_in,
                              void* d_out, int out_size, void* d_ws, size_t ws_size,
                              hipStream_t stream) {
  (void)in_sizes; (void)n_in; (void)out_size; (void)ws_size;
  const float* x   = (const float*)d_in[0];
  const float* W1  = (const float*)d_in[1];
  const float* b1  = (const float*)d_in[2];
  const float* a1  = (const float*)d_in[3];
  const float* ba1 = (const float*)d_in[4];
  const float* a2  = (const float*)d_in[5];
  const float* ba2 = (const float*)d_in[6];
  float* out = (float*)d_out;

  char* ws = (char*)d_ws;
  float*    sf   = (float*)(ws);                                   // 2 MB
  _Float16* Bpk  = (_Float16*)(ws + (size_t)(2u << 20));           // 2 MB
  char*     sc   = ws + (size_t)(4u << 20);
  float* f1   = (float*)(sc);
  float* f2   = (float*)(sc + 1 * 32768);
  float* E1   = (float*)(sc + 2 * 32768);
  float* E2   = (float*)(sc + 3 * 32768);
  float* m    = (float*)(sc + 4 * 32768);
  float* Z    = (float*)(sc + 5 * 32768);
  float* pmax = (float*)(sc + 6 * 32768);
  float* Wc   = (float*)(sc + 6 * 32768 + 256);

  k1_seqfts<<<dim3(N_NODES / 16), dim3(128), 0, stream>>>(x, W1, b1, sf);
  k2_fts<<<dim3(N_NODES / 256), dim3(256), 0, stream>>>(sf, a1, ba1, a2, ba2, f1, f2, pmax);
  k3_prep<<<dim3(1), dim3(256), 0, stream>>>(pmax, f1, f2, m, E1, E2, Wc);
  k4_z<<<dim3(N_NODES / 256), dim3(256), 0, stream>>>(f1, E1, E2, f2, m, Z);
  k5_pack<<<dim3((N_NODES * O_OUT) / 256), dim3(256), 0, stream>>>(sf, f2, m, Z, Bpk, Wc);
  k6_attn<<<dim3(N_NODES / 128), dim3(256), 0, stream>>>(Bpk, f1, f2, E1, E2, Wc, out);
}
